// SliceAttention_18983755449035
// MI455X (gfx1250) — matrix operand order repaired and checked by reading, not compiled, not run on hardware
//
#include <hip/hip_runtime.h>
#include <hip/hip_bf16.h>
#include <math.h>

typedef _Float16 f16;
typedef __attribute__((ext_vector_type(16))) _Float16 v16h;
typedef __attribute__((ext_vector_type(8)))  float    v8f;

#define B_   4
#define N_   16384
#define HID_ 512
#define H_   8
#define M_   32
#define D_   64
#define BN_  (B_ * N_)

// ---- d_out layout (float32, concatenated tuple outputs) ----
#define OUT_OFF   0
#define SW_OFF    ((size_t)BN_ * HID_)                       // 33,554,432
#define ALPHA_OFF (SW_OFF + (size_t)B_ * H_ * M_ * N_)       // 50,331,648
#define ZEROS_OFF (ALPHA_OFF + H_)                           // 50,331,656
#define ATTN_OFF  (ZEROS_OFF + H_ * M_)                      // 50,331,912

// ---------------- WMMA fragment helpers (CDNA5 16x16x32 f16) ----------------
// A (16xK slab, 16-bit): lane m = l&15 holds row m; half-select hs = l>>4;
// elements 0..15 -> K = k0 + hs*16 + e
static __device__ inline v8f zero8() {
  v8f z;
#pragma unroll
  for (int i = 0; i < 8; ++i) z[i] = 0.f;
  return z;
}

static __device__ inline v16h load_a_f16(const f16* p, int ld, int lane) {
  const int r = lane & 15, hs = lane >> 4;
  const f16* q = p + (size_t)r * ld + hs * 16;
  v16h a;
#pragma unroll
  for (int e = 0; e < 8; ++e) a[e] = q[e];
#pragma unroll
  for (int e = 0; e < 8; ++e) a[e + 8] = q[e + 8];
  return a;
}

static __device__ inline v16h load_a_cvt_f32(const float* p, int ld, int lane) {
  const int r = lane & 15, hs = lane >> 4;
  const float* q = p + (size_t)r * ld + hs * 16;
  v16h a;
#pragma unroll
  for (int e = 0; e < 8; ++e) a[e] = (f16)q[e];
#pragma unroll
  for (int e = 0; e < 8; ++e) a[e + 8] = (f16)q[e + 8];
  return a;
}

// B (Kx16, 16-bit), representing W^T with row-major W (row = out col, ld = K dim):
// lane col n = l&15, holds K = k0 + (l>>4)*16 + e, contiguous in memory.
static __device__ inline v16h load_b_f16(const f16* p /* at (col0,k0) in W */, int ld, int lane) {
  const f16* q = p + (size_t)(lane & 15) * ld + (lane >> 4) * 16;
  v16h b;
#pragma unroll
  for (int e = 0; e < 16; ++e) b[e] = q[e];
  return b;
}

static __device__ inline v8f wmma16(v16h a, v16h b, v8f c) {
  return __builtin_amdgcn_wmma_f32_16x16x32_f16(false, a, false, b, (short)0, c, false, false);
}

// ---------------- kernel 1: weight conversion to f16 ----------------
#define WKV_E  (2 * HID_ * HID_)   // 524288
#define WOUT_E (HID_ * HID_)       // 262144
#define WTQ_E  (H_ * M_ * D_)      // 16384
#define MIX_E  (H_ * M_ * H_ * M_) // 65536
#define CVT_TOT (WKV_E + WOUT_E + WTQ_E + MIX_E)

__global__ void k_cvt(const float* wkv, const float* wout, const float* wtq, const float* mix,
                      f16* wkv16, f16* wout16, f16* wtq16, f16* mix16) {
  int i = blockIdx.x * blockDim.x + threadIdx.x;
  if (i >= CVT_TOT) return;
  if (i < WKV_E) { wkv16[i] = (f16)wkv[i]; return; }
  i -= WKV_E;
  if (i < WOUT_E) { wout16[i] = (f16)wout[i]; return; }
  i -= WOUT_E;
  if (i < WTQ_E) { wtq16[i] = (f16)wtq[i]; return; }
  i -= WTQ_E;
  mix16[i] = (f16)mix[i];
}

// ---------------- kernel 0: init accumulators + tuple constants ----------------
#define ACC_E (B_ * H_ * M_ * D_)  // 65536
#define DEN_E (B_ * H_ * M_)       // 1024
#define INIT_TOT (ACC_E + DEN_E + H_ + H_ * M_)

__global__ void k_init(float* tok_acc, float* denom, const float* alpha1, float* out_base) {
  int i = blockIdx.x * blockDim.x + threadIdx.x;
  if (i >= INIT_TOT) return;
  if (i < ACC_E) { tok_acc[i] = 0.f; return; }
  i -= ACC_E;
  if (i < DEN_E) { denom[i] = 0.f; return; }
  i -= DEN_E;
  if (i < H_) { out_base[ALPHA_OFF + i] = alpha1[i]; return; }
  i -= H_;
  out_base[ZEROS_OFF + i] = 0.f;
}

// ---------------- kernel 2: kv = x @ wkv^T + bkv  (WMMA) ----------------
// grid 2048 (32 tokens each), block 256 = 8 waves; wave w -> cols [128w,128w+128).
// Two 16-row A fragments per wave: each B fragment is reused for 2 WMMAs.
__global__ __launch_bounds__(256) void k_kv_gemm(const float* __restrict__ x,
                                                 const f16* __restrict__ wkv16,
                                                 const float* __restrict__ bkv,
                                                 f16* __restrict__ xk16, f16* __restrict__ xv16) {
  const int lane = threadIdx.x & 31, wave = threadIdx.x >> 5;
  const int row0 = blockIdx.x * 32;
  v8f acc0[8], acc1[8];
#pragma unroll
  for (int c = 0; c < 8; ++c) { acc0[c] = zero8(); acc1[c] = zero8(); }
  const float* xrow0 = x + (size_t)row0 * HID_;
  const float* xrow1 = x + (size_t)(row0 + 16) * HID_;
  for (int ks = 0; ks < 16; ++ks) {
    v16h a0 = load_a_cvt_f32(xrow0 + ks * 32, HID_, lane);
    v16h a1 = load_a_cvt_f32(xrow1 + ks * 32, HID_, lane);
    if (ks + 1 < 16) {
      __builtin_prefetch(xrow0 + (ks + 1) * 32 + (size_t)(lane & 15) * HID_, 0, 3);
      __builtin_prefetch(xrow1 + (ks + 1) * 32 + (size_t)(lane & 15) * HID_, 0, 3);
    }
#pragma unroll
    for (int c = 0; c < 8; ++c) {
      const int col0 = wave * 128 + c * 16;
      v16h b = load_b_f16(wkv16 + (size_t)col0 * HID_ + ks * 32, HID_, lane);
      acc0[c] = wmma16(a0, b, acc0[c]);
      acc1[c] = wmma16(a1, b, acc1[c]);
    }
  }
  const int hs = lane >> 4, cl = lane & 15;
#pragma unroll
  for (int c = 0; c < 8; ++c) {
    const int col = wave * 128 + c * 16 + cl;
    const float bias = bkv[col];
#pragma unroll
    for (int g = 0; g < 2; ++g) {
#pragma unroll
      for (int r = 0; r < 8; ++r) {
        const int row = row0 + g * 16 + r + hs * 8;
        const float v = (g ? acc1[c][r] : acc0[c][r]) + bias;
        if (col < HID_) xk16[(size_t)row * HID_ + col] = (f16)v;
        else            xv16[(size_t)row * HID_ + (col - HID_)] = (f16)v;
      }
    }
  }
}

// ---------------- kernel 3: slice scores + mix + softmax -> sw ----------------
// grid (N/16, B), block 128 = 4 waves. One 16-token tile per block.
__global__ __launch_bounds__(128) void k_scores(const f16* __restrict__ xk16,
                                                const f16* __restrict__ wtq16,
                                                const f16* __restrict__ mix16,
                                                const float* __restrict__ alpha1,
                                                float* __restrict__ out_base) {
  __shared__ f16  s0[16][272];   // 16 tokens x 256 channels (h*32+m), padded
  __shared__ float s1[16][272];  // mixed scores
  const int lane = threadIdx.x & 31, wave = threadIdx.x >> 5;
  const int b = blockIdx.y, n0 = blockIdx.x * 16;
  const int hs = lane >> 4, cl = lane & 15;

  // stage A: s0[c,n] = wtq . xk  (per-head K=64 GEMM)
  for (int hh = 0; hh < 2; ++hh) {
    const int h = wave * 2 + hh;
    const f16* arow = xk16 + (size_t)(b * N_ + n0) * HID_ + h * 64;
#pragma unroll
    for (int mt = 0; mt < 2; ++mt) {
      v8f acc = zero8();
#pragma unroll
      for (int ks = 0; ks < 2; ++ks) {
        v16h a = load_a_f16(arow + ks * 32, HID_, lane);
        v16h bb = load_b_f16(wtq16 + (size_t)(h * 32 + mt * 16) * 64 + ks * 32, 64, lane);
        acc = wmma16(a, bb, acc);
      }
#pragma unroll
      for (int r = 0; r < 8; ++r)
        s0[r + hs * 8][h * 32 + mt * 16 + cl] = (f16)acc[r];
    }
  }
  __syncthreads();

  // stage B: s1 = mix1_w @ s0  (K=256), A fragments from LDS
  for (int q = 0; q < 4; ++q) {
    const int ot = wave + q * 4;
    v8f acc = zero8();
#pragma unroll
    for (int kt = 0; kt < 8; ++kt) {
      v16h a = load_a_f16(&s0[0][0] + kt * 32, 272, lane);
      v16h bb = load_b_f16(mix16 + (size_t)(ot * 16) * 256 + kt * 32, 256, lane);
      acc = wmma16(a, bb, acc);
    }
#pragma unroll
    for (int r = 0; r < 8; ++r)
      s1[r + hs * 8][ot * 16 + cl] = acc[r];
  }
  __syncthreads();

  // softmax over M within each head; thread t -> (token = t&15, head = t>>4)
  const int tok = threadIdx.x & 15, h = threadIdx.x >> 4;
  const float al = alpha1[h];
  float mx = -3.4e38f;
#pragma unroll
  for (int m = 0; m < M_; ++m) {
    float y = s1[tok][h * 32 + m] * al;
    mx = fmaxf(mx, y);
  }
  float sum = 0.f;
#pragma unroll
  for (int m = 0; m < M_; ++m)
    sum += expf(s1[tok][h * 32 + m] * al - mx);
  const float inv = 1.f / sum;
  float* swp = out_base + SW_OFF + (size_t)((b * H_ + h) * M_) * N_ + n0 + tok;
#pragma unroll
  for (int m = 0; m < M_; ++m)
    swp[(size_t)m * N_] = expf(s1[tok][h * 32 + m] * al - mx) * inv;
}

// ---------------- kernel 4: tok_acc += sw . xv ; denom += sum(sw) ----------------
// grid (16, B*H): each block reduces 1024 tokens for one (b,h)
__global__ __launch_bounds__(256) void k_tok_reduce(const float* __restrict__ out_base,
                                                    const f16* __restrict__ xv16,
                                                    float* __restrict__ tok_acc,
                                                    float* __restrict__ denom) {
  const int bh = blockIdx.y, b = bh >> 3, h = bh & 7;
  const int m = threadIdx.x >> 3, d0 = (threadIdx.x & 7) * 8;
  const int nbase = blockIdx.x * 1024;
  const float* swp = out_base + SW_OFF + (size_t)(bh * M_ + m) * N_ + nbase;
  const f16* xvp = xv16 + (size_t)(b * N_ + nbase) * HID_ + h * 64 + d0;
  float acc[8];
#pragma unroll
  for (int j = 0; j < 8; ++j) acc[j] = 0.f;
  float dsum = 0.f;
  for (int i = 0; i < 1024; ++i) {
    const float s = swp[i];
    const f16* xp = xvp + (size_t)i * HID_;
#pragma unroll
    for (int j = 0; j < 8; ++j) acc[j] += s * (float)xp[j];
    dsum += s;
  }
  float* dst = tok_acc + (size_t)(bh * M_ + m) * 64 + d0;
#pragma unroll
  for (int j = 0; j < 8; ++j) atomicAdd(&dst[j], acc[j]);
  if ((threadIdx.x & 7) == 0) atomicAdd(&denom[bh * M_ + m], dsum);
}

// ---------------- kernel 5a: normalize + layernorm -> tokln [B][M][HID] ----------------
__global__ __launch_bounds__(256) void k_tok_ln(const float* __restrict__ tok_acc,
                                                const float* __restrict__ denom,
                                                const float* __restrict__ lnw,
                                                const float* __restrict__ lnb,
                                                float* __restrict__ tokln) {
  const int b = blockIdx.x, t = threadIdx.x;
  const int h = t >> 5, m = t & 31;
  const float* src = tok_acc + (size_t)(b * 256 + t) * 64;
  const float invden = 1.f / (denom[b * 256 + t] + 1e-5f);
  float s = 0.f, s2 = 0.f;
  for (int d = 0; d < 64; ++d) {
    const float v = src[d] * invden;
    s += v; s2 += v * v;
  }
  const float mu = s * (1.f / 64.f);
  const float var = s2 * (1.f / 64.f) - mu * mu;
  const float rs = rsqrtf(var + 1e-5f);
  float* dst = tokln + (size_t)(b * M_ + m) * HID_ + h * 64;
  for (int d = 0; d < 64; ++d)
    dst[d] = (src[d] * invden - mu) * rs * lnw[d] + lnb[d];
}

// ---------------- kernel 5b: qkv = tokln @ wqkv^T ----------------
__global__ __launch_bounds__(256) void k_qkv(const float* __restrict__ tokln,
                                             const float* __restrict__ wqkv,
                                             float* __restrict__ qkvws) {
  const int gid = blockIdx.x * blockDim.x + threadIdx.x;
  if (gid >= B_ * M_ * 3 * HID_) return;
  const int b = gid / (M_ * 3 * HID_);
  const int r = gid % (M_ * 3 * HID_);
  const int m = r / (3 * HID_), o = r % (3 * HID_);
  const float* a = tokln + (size_t)(b * M_ + m) * HID_;
  const float* w = wqkv + (size_t)o * HID_;
  float s = 0.f;
  for (int c = 0; c < HID_; ++c) s += a[c] * w[c];
  qkvws[gid] = s;
}

// ---------------- kernel 5c: inner 32x32 MHA ----------------
__global__ __launch_bounds__(256) void k_inner_attn(const float* __restrict__ qkvws,
                                                    float* __restrict__ out_base,
                                                    float* __restrict__ out_tok) {
  __shared__ float qs[32][64], ks[32][64], vs[32][64], dots[32][33];
  const int bh = blockIdx.x, b = bh >> 3, h = bh & 7, t = threadIdx.x;
#pragma unroll
  for (int j = 0; j < 8; ++j) {
    const int idx = t * 8 + j, m = idx >> 6, d = idx & 63;
    const size_t base = (size_t)(b * M_ + m) * (3 * HID_) + h * 64 + d;
    qs[m][d] = qkvws[base];
    ks[m][d] = qkvws[base + HID_];
    vs[m][d] = qkvws[base + 2 * HID_];
  }
  __syncthreads();
#pragma unroll
  for (int j = 0; j < 4; ++j) {
    const int idx = t * 4 + j, qi = idx >> 5, ki = idx & 31;
    float s = 0.f;
    for (int d = 0; d < 64; ++d) s += qs[qi][d] * ks[ki][d];
    dots[qi][ki] = s * 0.125f;
  }
  __syncthreads();
  if (t < 32) {
    float mx = -3.4e38f;
    for (int k = 0; k < 32; ++k) mx = fmaxf(mx, dots[t][k]);
    float sum = 0.f;
    for (int k = 0; k < 32; ++k) sum += expf(dots[t][k] - mx);
    const float inv = 1.f / sum;
    float* ap = out_base + ATTN_OFF + (size_t)(bh * M_ + t) * M_;
    for (int k = 0; k < 32; ++k) {
      const float a = expf(dots[t][k] - mx) * inv;
      dots[t][k] = a;
      ap[k] = a;
    }
  }
  __syncthreads();
#pragma unroll
  for (int j = 0; j < 8; ++j) {
    const int idx = t * 8 + j, m = idx >> 6, d = idx & 63;
    float s = 0.f;
    for (int k = 0; k < 32; ++k) s += dots[m][k] * vs[k][d];
    out_tok[(size_t)(bh * M_ + m) * 64 + d] = s;
  }
}

// ---------------- kernel 6: out0[n, h*64+d] = sum_m out_tok[m,d]*sw[m,n] ----------------
// grid (N/256, B*H), block 256
__global__ __launch_bounds__(256) void k_broadcast(const float* __restrict__ out_base,
                                                   const float* __restrict__ out_tok,
                                                   f16* __restrict__ out0) {
  __shared__ float ot[32][64];
  const int bh = blockIdx.y, b = bh >> 3, h = bh & 7, t = threadIdx.x;
#pragma unroll
  for (int j = 0; j < 8; ++j) {
    const int idx = t * 8 + j;
    ot[idx >> 6][idx & 63] = out_tok[(size_t)bh * 2048 + idx];
  }
  __syncthreads();
  const int n = blockIdx.x * 256 + t;
  float acc[64];
#pragma unroll
  for (int d = 0; d < 64; ++d) acc[d] = 0.f;
  const float* swp = out_base + SW_OFF + (size_t)(bh * M_) * N_ + n;
  for (int m = 0; m < 32; ++m) {
    const float s = swp[(size_t)m * N_];
#pragma unroll
    for (int d = 0; d < 64; ++d) acc[d] += s * ot[m][d];
  }
  f16* dst = out0 + (size_t)(b * N_ + n) * HID_ + h * 64;
#pragma unroll
  for (int d = 0; d < 64; ++d) dst[d] = (f16)acc[d];
}

// ---------------- kernel 7: out = out0 @ wout^T + bout  (WMMA) ----------------
// grid 2048 (32 rows), block 256 = 8 waves; wave w -> cols [64w, 64w+64).
// Two A fragments per wave: each B fragment feeds 2 WMMAs.
__global__ __launch_bounds__(256) void k_out_gemm(const f16* __restrict__ out0,
                                                  const f16* __restrict__ wout16,
                                                  const float* __restrict__ bout,
                                                  float* __restrict__ out_base) {
  const int lane = threadIdx.x & 31, wave = threadIdx.x >> 5;
  const int row0 = blockIdx.x * 32;
  v8f acc0[4], acc1[4];
#pragma unroll
  for (int c = 0; c < 4; ++c) { acc0[c] = zero8(); acc1[c] = zero8(); }
  const f16* arow0 = out0 + (size_t)row0 * HID_;
  const f16* arow1 = out0 + (size_t)(row0 + 16) * HID_;
  for (int ks = 0; ks < 16; ++ks) {
    v16h a0 = load_a_f16(arow0 + ks * 32, HID_, lane);
    v16h a1 = load_a_f16(arow1 + ks * 32, HID_, lane);
    if (ks + 1 < 16) {
      __builtin_prefetch(arow0 + (ks + 1) * 32 + (size_t)(lane & 15) * HID_, 0, 3);
      __builtin_prefetch(arow1 + (ks + 1) * 32 + (size_t)(lane & 15) * HID_, 0, 3);
    }
#pragma unroll
    for (int c = 0; c < 4; ++c) {
      const int col0 = wave * 64 + c * 16;
      v16h b = load_b_f16(wout16 + (size_t)col0 * HID_ + ks * 32, HID_, lane);
      acc0[c] = wmma16(a0, b, acc0[c]);
      acc1[c] = wmma16(a1, b, acc1[c]);
    }
  }
  const int hs = lane >> 4, cl = lane & 15;
#pragma unroll
  for (int c = 0; c < 4; ++c) {
    const int col = wave * 64 + c * 16 + cl;
    const float bias = bout[col];
#pragma unroll
    for (int r = 0; r < 8; ++r) {
      out_base[OUT_OFF + (size_t)(row0 + r + hs * 8) * HID_ + col] = acc0[c][r] + bias;
      out_base[OUT_OFF + (size_t)(row0 + 16 + r + hs * 8) * HID_ + col] = acc1[c][r] + bias;
    }
  }
}

// ---------------- host launcher ----------------
extern "C" void kernel_launch(void* const* d_in, const int* in_sizes, int n_in,
                              void* d_out, int out_size, void* d_ws, size_t ws_size,
                              hipStream_t stream) {
  (void)in_sizes; (void)n_in; (void)out_size; (void)ws_size;
  const float* x      = (const float*)d_in[0];
  const float* wkv    = (const float*)d_in[1];
  const float* bkv    = (const float*)d_in[2];
  const float* wtq    = (const float*)d_in[3];
  const float* alpha1 = (const float*)d_in[4];
  const float* mix1w  = (const float*)d_in[5];
  const float* ln1w   = (const float*)d_in[6];
  const float* ln1b   = (const float*)d_in[7];
  const float* wqkv   = (const float*)d_in[8];
  const float* wout   = (const float*)d_in[9];
  const float* bout   = (const float*)d_in[10];
  float* outb = (float*)d_out;

  // workspace carve-up (all 256B multiples)
  char* ws = (char*)d_ws;
  size_t o = 0;
  f16* wkv16  = (f16*)(ws + o); o += (size_t)WKV_E * 2;            // 1 MB
  f16* wout16 = (f16*)(ws + o); o += (size_t)WOUT_E * 2;           // 0.5 MB
  f16* wtq16  = (f16*)(ws + o); o += (size_t)WTQ_E * 2;
  f16* mix16  = (f16*)(ws + o); o += (size_t)MIX_E * 2;
  f16* xk16   = (f16*)(ws + o); o += (size_t)BN_ * HID_ * 2;       // 64 MB (reused as out0)
  f16* xv16   = (f16*)(ws + o); o += (size_t)BN_ * HID_ * 2;       // 64 MB
  float* tok_acc = (float*)(ws + o); o += (size_t)ACC_E * 4;
  float* denom   = (float*)(ws + o); o += (size_t)DEN_E * 4;
  float* tokln   = (float*)(ws + o); o += (size_t)B_ * M_ * HID_ * 4;
  float* qkvws   = (float*)(ws + o); o += (size_t)B_ * M_ * 3 * HID_ * 4;
  float* out_tok = (float*)(ws + o); o += (size_t)B_ * H_ * M_ * D_ * 4;
  f16* out0 = xk16;  // alias: xk dead after k_scores, out0 produced after

  k_cvt<<<(CVT_TOT + 255) / 256, 256, 0, stream>>>(wkv, wout, wtq, mix1w,
                                                   wkv16, wout16, wtq16, mix16);
  k_init<<<(INIT_TOT + 255) / 256, 256, 0, stream>>>(tok_acc, denom, alpha1, outb);
  k_kv_gemm<<<BN_ / 32, 256, 0, stream>>>(x, wkv16, bkv, xk16, xv16);
  k_scores<<<dim3(N_ / 16, B_), 128, 0, stream>>>(xk16, wtq16, mix16, alpha1, outb);
  k_tok_reduce<<<dim3(16, B_ * H_), 256, 0, stream>>>(outb, xv16, tok_acc, denom);
  k_tok_ln<<<B_, 256, 0, stream>>>(tok_acc, denom, ln1w, ln1b, tokln);
  k_qkv<<<(B_ * M_ * 3 * HID_ + 255) / 256, 256, 0, stream>>>(tokln, wqkv, qkvws);
  k_inner_attn<<<B_ * H_, 256, 0, stream>>>(qkvws, outb, out_tok);
  k_broadcast<<<dim3(N_ / 256, B_ * H_), 256, 0, stream>>>(outb, out_tok, out0);
  k_out_gemm<<<BN_ / 32, 256, 0, stream>>>(out0, wout16, bout, outb);
}
